// GConv2d_75892072120363
// MI455X (gfx1250) — matrix operand order repaired and checked by reading, not compiled, not run on hardware
//
#include <hip/hip_runtime.h>
#include <math.h>

// ---------------------------------------------------------------------------
// GConv2d on MI455X (gfx1250): implicit-GEMM conv via bf16 split (hi+lo) WMMA.
//   Y[b] (256 x 4096) = W[b] (256 x 2304) * im2col(x[b]) (2304 x 4096)
// fp32 emulated with 3 bf16 WMMAs per K=32 chunk (a_hi*b_hi + a_hi*b_lo + a_lo*b_hi).
// A-tile staged with GLOBAL_LOAD_ASYNC_TO_LDS (ASYNCcnt) when available.
// ---------------------------------------------------------------------------

typedef __attribute__((ext_vector_type(16))) __bf16 v16bf;
typedef __attribute__((ext_vector_type(8)))  __bf16 bf16x8;
typedef __attribute__((ext_vector_type(8)))  float  v8f;
typedef __attribute__((vector_size(16)))     int    vi4;   // matches builtin param

#define BATCH   16
#define CTOT    256          // R*Cin
#define HW      4096         // 64*64
#define MDIM    256          // R*O out channels
#define KDIM    2304         // CTOT*9
#define KCHUNKS 72           // KDIM/32
#define MTILE   128
#define NTILE   128
#define ASTR    40           // padded LDS row stride (halfs) for A  (80 B)
#define BSTR    40           // padded LDS row stride (halfs) for B^T (80 B)

#define AS1 __attribute__((address_space(1)))
#define AS3 __attribute__((address_space(3)))

#if defined(__has_builtin)
#  if __has_builtin(__builtin_amdgcn_global_load_async_to_lds_b128)
#    define USE_ASYNC_LDS 1
#  endif
#  if __has_builtin(__builtin_amdgcn_s_wait_asynccnt)
#    define HAVE_WAIT_ASYNC 1
#  endif
#endif

__device__ __forceinline__ unsigned short f32_to_bf16(float f) {
  unsigned int u = __float_as_uint(f);
  u += 0x7FFFu + ((u >> 16) & 1u);          // round-to-nearest-even
  return (unsigned short)(u >> 16);
}
__device__ __forceinline__ float bf16_to_f32(unsigned short h) {
  return __uint_as_float(((unsigned int)h) << 16);
}

// ---------------- 1) global average pool: one block per (b,c) image --------
__global__ __launch_bounds__(256) void pool_kernel(const float* __restrict__ x,
                                                   float* __restrict__ pooled) {
  __shared__ float red[256];
  const float* p = x + (size_t)blockIdx.x * HW;
  float s = 0.f;
#pragma unroll
  for (int i = 0; i < 16; ++i) s += p[threadIdx.x + (i << 8)];
  red[threadIdx.x] = s;
  __syncthreads();
  for (int off = 128; off > 0; off >>= 1) {
    if (threadIdx.x < off) red[threadIdx.x] += red[threadIdx.x + off];
    __syncthreads();
  }
  if (threadIdx.x == 0) pooled[blockIdx.x] = red[0] * (1.0f / 4096.0f);
}

// ---------------- 2) coeffs = sigmoid(pooled @ fc_w.T + fc_b) --------------
__global__ void coeff_kernel(const float* __restrict__ pooled,
                             const float* __restrict__ fc_w,
                             const float* __restrict__ fc_b,
                             float* __restrict__ coeffs) {
  int t = threadIdx.x;
  if (t >= BATCH * 4) return;
  int b = t >> 2, k = t & 3;
  float z = fc_b[k];
  for (int c = 0; c < CTOT; ++c) z += pooled[b * CTOT + c] * fc_w[k * CTOT + c];
  coeffs[b * 4 + k] = 1.0f / (1.0f + __expf(-z));
}

// ---------------- 3) build per-batch weights, split into bf16 hi/lo --------
// W[b][m][kd], m = ro*64+o, kd = (ri*64+c)*9 + fi*3 + fj, rotated by (ro-ri)%4.
__global__ __launch_bounds__(256) void build_w_kernel(
    const float* __restrict__ bank,      // (4, 64, 64, 3, 3)
    const float* __restrict__ coeffs,    // (16, 4)
    unsigned short* __restrict__ Ahi,
    unsigned short* __restrict__ Alo) {
  int idx = blockIdx.x * 256 + threadIdx.x;
  const int total = BATCH * MDIM * KDIM;
  if (idx >= total) return;
  int b   = idx / (MDIM * KDIM);
  int rem = idx % (MDIM * KDIM);
  int m   = rem / KDIM;
  int kd  = rem % KDIM;
  int ro = m >> 6, o = m & 63;
  int cit = kd / 9, r9 = kd % 9;
  int fi = r9 / 3, fj = r9 % 3;
  int ri = cit >> 6, c = cit & 63;
  int rr = (ro - ri) & 3;
  int si, sj;
  switch (rr) {                 // rot90 applied rr times (counterclockwise)
    case 0:  si = fi;     sj = fj;     break;
    case 1:  si = fj;     sj = 2 - fi; break;
    case 2:  si = 2 - fi; sj = 2 - fj; break;
    default: si = 2 - fj; sj = fi;     break;
  }
  float v = 0.f;
#pragma unroll
  for (int k = 0; k < 4; ++k)
    v += coeffs[b * 4 + k] * bank[(((k * 64 + o) * 64 + c) * 3 + si) * 3 + sj];
  unsigned short h = f32_to_bf16(v);
  Ahi[idx] = h;
  Alo[idx] = f32_to_bf16(v - bf16_to_f32(h));
}

// ---------------- 4) implicit-GEMM conv with bf16-split WMMA ---------------
// Grid: (N tiles = 32, M tiles = 2, batch = 16). 256 threads = 8 wave32.
// Wave w owns M-rows [w*16, w*16+16) of the 128-row tile, all 128 N-cols
// -> 8 f32 accumulators, 24 WMMAs per K-chunk.
__global__ __launch_bounds__(256) void conv_wmma_kernel(
    const float* __restrict__ x,
    const unsigned short* __restrict__ Ahi,
    const unsigned short* __restrict__ Alo,
    float* __restrict__ out) {
  __shared__ unsigned short sAh[MTILE * ASTR];
  __shared__ unsigned short sAl[MTILE * ASTR];
  __shared__ unsigned short sBh[NTILE * BSTR];   // B^T: [n][k], k contiguous
  __shared__ unsigned short sBl[NTILE * BSTR];

  const int b     = blockIdx.z;
  const int mBase = blockIdx.y * MTILE;
  const int nBase = blockIdx.x * NTILE;
  const int tid   = threadIdx.x;
  const int lane  = tid & 31;
  const int wave  = tid >> 5;                    // 0..7 -> M sub-strip

  v8f acc[8] = {v8f{}, v8f{}, v8f{}, v8f{}, v8f{}, v8f{}, v8f{}, v8f{}};

  // A-tile fill mapping: row = tid>>1 (0..127), 32-byte chunk = tid&1
  const int arow   = tid >> 1;
  const int achunk = (tid & 1) * 16;                      // halfs
  const size_t abase = ((size_t)b * MDIM + (mBase + arow)) * KDIM;
  const int aldsOff = arow * ASTR + achunk;               // halfs

  // B-tile fill mapping: k-slice per wave, n across lanes (coalesced x loads)
  const int kSlice = tid >> 5;                            // 0..7
  const int nLane  = tid & 31;                            // 0..31

  // Fragment addressing (ISA 16-bit 16x32 A / 32x16 B layouts, wave32)
  const int fl = lane & 15;
  const int ah = lane >> 4;                               // half select
  const unsigned short* aRowH = sAh + (wave * 16 + fl) * ASTR + ah * 8;
  const unsigned short* aRowL = sAl + (wave * 16 + fl) * ASTR + ah * 8;

  for (int kc = 0; kc < KCHUNKS; ++kc) {
    // ---- stage A tile (hi/lo): async global->LDS when available ----
    {
      const size_t g = abase + kc * 32 + achunk;
#if defined(USE_ASYNC_LDS)
      __builtin_amdgcn_global_load_async_to_lds_b128(
          (AS1 vi4*)(Ahi + g),     (AS3 vi4*)(sAh + aldsOff),     0, 0);
      __builtin_amdgcn_global_load_async_to_lds_b128(
          (AS1 vi4*)(Ahi + g + 8), (AS3 vi4*)(sAh + aldsOff + 8), 0, 0);
      __builtin_amdgcn_global_load_async_to_lds_b128(
          (AS1 vi4*)(Alo + g),     (AS3 vi4*)(sAl + aldsOff),     0, 0);
      __builtin_amdgcn_global_load_async_to_lds_b128(
          (AS1 vi4*)(Alo + g + 8), (AS3 vi4*)(sAl + aldsOff + 8), 0, 0);
#else
      *(bf16x8*)(sAh + aldsOff)     = *(const bf16x8*)(Ahi + g);
      *(bf16x8*)(sAh + aldsOff + 8) = *(const bf16x8*)(Ahi + g + 8);
      *(bf16x8*)(sAl + aldsOff)     = *(const bf16x8*)(Alo + g);
      *(bf16x8*)(sAl + aldsOff + 8) = *(const bf16x8*)(Alo + g + 8);
#endif
      if (kc + 1 < KCHUNKS) {                 // global_prefetch_b8 for next chunk
        __builtin_prefetch(Ahi + g + 32, 0, 1);
        __builtin_prefetch(Alo + g + 32, 0, 1);
      }
    }
    // ---- stage B^T tile: on-the-fly im2col with zero padding ----
#pragma unroll
    for (int ko = 0; ko < 32; ko += 8) {
      const int k  = kSlice + ko;
      const int kk = kc * 32 + k;
      const int ci = kk / 9;
      const int r9 = kk - ci * 9;
      const int fi = r9 / 3, fj = r9 % 3;
      const size_t cbase = ((size_t)b * CTOT + ci) << 12;
#pragma unroll
      for (int no = 0; no < NTILE; no += 32) {
        const int n = nLane + no;
        const int p = nBase + n;
        const int yy = (p >> 6) + fi - 1;
        const int xx = (p & 63) + fj - 1;
        float v = 0.f;
        if ((unsigned)yy < 64u && (unsigned)xx < 64u)
          v = x[cbase + (yy << 6) + xx];
        unsigned short h = f32_to_bf16(v);
        sBh[n * BSTR + k] = h;
        sBl[n * BSTR + k] = f32_to_bf16(v - bf16_to_f32(h));
      }
    }
#if defined(USE_ASYNC_LDS)
#  if defined(HAVE_WAIT_ASYNC)
    __builtin_amdgcn_s_wait_asynccnt(0);
#  else
    asm volatile("s_wait_asynccnt 0" ::: "memory");
#  endif
#endif
    __syncthreads();

    // ---- A fragments (two 16B chunks -> v16bf per ISA layout) ----
    bf16x8 a0 = *(const bf16x8*)(aRowH);
    bf16x8 a1 = *(const bf16x8*)(aRowH + 16);
    v16bf aH  = __builtin_shufflevector(a0, a1, 0,1,2,3,4,5,6,7,8,9,10,11,12,13,14,15);
    bf16x8 l0 = *(const bf16x8*)(aRowL);
    bf16x8 l1 = *(const bf16x8*)(aRowL + 16);
    v16bf aL  = __builtin_shufflevector(l0, l1, 0,1,2,3,4,5,6,7,8,9,10,11,12,13,14,15);

#pragma unroll
    for (int s = 0; s < 8; ++s) {
      const int nn = (s * 16 + fl) * BSTR + ah * 8;
      bf16x8 b0 = *(const bf16x8*)(sBh + nn);
      bf16x8 b1 = *(const bf16x8*)(sBh + nn + 16);
      v16bf bH  = __builtin_shufflevector(b0, b1, 0,1,2,3,4,5,6,7,8,9,10,11,12,13,14,15);
      bf16x8 c0 = *(const bf16x8*)(sBl + nn);
      bf16x8 c1 = *(const bf16x8*)(sBl + nn + 16);
      v16bf bL  = __builtin_shufflevector(c0, c1, 0,1,2,3,4,5,6,7,8,9,10,11,12,13,14,15);

      acc[s] = __builtin_amdgcn_wmma_f32_16x16x32_bf16(false, aH, false, bH,
                                                       (short)0, acc[s], false, false);
      acc[s] = __builtin_amdgcn_wmma_f32_16x16x32_bf16(false, aH, false, bL,
                                                       (short)0, acc[s], false, false);
      acc[s] = __builtin_amdgcn_wmma_f32_16x16x32_bf16(false, aL, false, bH,
                                                       (short)0, acc[s], false, false);
    }
    __syncthreads();
  }

  // ---- write out: C/D layout (VGPR r: lanes0-15 M=r, lanes16-31 M=r+8) ----
#pragma unroll
  for (int s = 0; s < 8; ++s) {
    const int ncol = nBase + s * 16 + fl;
#pragma unroll
    for (int r = 0; r < 8; ++r) {
      const int mrow = mBase + wave * 16 + r + ah * 8;
      out[(((size_t)b * MDIM + mrow) << 12) + ncol] = acc[s][r];
    }
  }
}

// ---------------------------------------------------------------------------
extern "C" void kernel_launch(void* const* d_in, const int* in_sizes, int n_in,
                              void* d_out, int out_size, void* d_ws, size_t ws_size,
                              hipStream_t stream) {
  (void)in_sizes; (void)n_in; (void)out_size; (void)ws_size;
  const float* x    = (const float*)d_in[0];   // (16, 256, 64, 64)
  const float* bank = (const float*)d_in[1];   // (4, 64, 64, 3, 3)
  const float* fc_w = (const float*)d_in[2];   // (4, 256)
  const float* fc_b = (const float*)d_in[3];   // (4,)
  float* out = (float*)d_out;                  // (16, 256, 64, 64)

  // workspace layout (all 16B aligned): pooled | coeffs | Ahi | Alo (~37.8 MB)
  float* pooled = (float*)d_ws;                          // 4096 f32
  float* coeffs = pooled + BATCH * CTOT;                 // 64 f32
  unsigned short* Ahi = (unsigned short*)(coeffs + 64);
  unsigned short* Alo = Ahi + (size_t)BATCH * MDIM * KDIM;

  pool_kernel<<<BATCH * CTOT, 256, 0, stream>>>(x, pooled);
  coeff_kernel<<<1, 64, 0, stream>>>(pooled, fc_w, fc_b, coeffs);

  const int totalW = BATCH * MDIM * KDIM;
  build_w_kernel<<<(totalW + 255) / 256, 256, 0, stream>>>(bank, coeffs, Ahi, Alo);

  dim3 grid(HW / NTILE, MDIM / MTILE, BATCH);  // (32, 2, 16)
  conv_wmma_kernel<<<grid, 256, 0, stream>>>(x, Ahi, Alo, out);
}
